// CGF_63634235458232
// MI455X (gfx1250) — matrix operand order repaired and checked by reading, not compiled, not run on hardware
//
#include <hip/hip_runtime.h>
#include <hip/hip_bf16.h>

// ---------------------------------------------------------------------------
// Tree-GRU for MI455X (gfx1250): bf16 WMMA 16x16x32, fp32 accum + fp32 gates.
// D=512, gates G=1536, arity A=8, levels 3 (64 nodes) -> 2 (8) -> 1 (1).
// ---------------------------------------------------------------------------

typedef __attribute__((ext_vector_type(16))) __bf16 v16bf;
typedef __attribute__((ext_vector_type(8)))  __bf16 v8bf;
typedef __attribute__((ext_vector_type(8)))  float  v8f;

#define DD 512
#define GG 1536

__device__ __forceinline__ unsigned short f2bf(float f) {
  unsigned int u = __float_as_uint(f);
  u += 0x7FFFu + ((u >> 16) & 1u);          // round-to-nearest-even
  return (unsigned short)(u >> 16);
}

// A-matrix fragment, 16x32 bf16 (M x K), row-major src with leading dim DD.
// ISA layout: lane l<16 -> M=l, K = kb..kb+7 (elems 0..7) and kb+16..kb+23
// (elems 8..15); lane l+16 -> same M, K offset +8 / +24.
__device__ __forceinline__ v16bf load_a_frag(const unsigned short* base,
                                             int mbase, int kbase, int lane) {
  int m  = mbase + (lane & 15);
  int ko = (lane >> 4) * 8;
  const unsigned short* p = base + m * DD + kbase + ko;
  v8bf c0 = *(const v8bf*)(p);
  v8bf c1 = *(const v8bf*)(p + 16);
  return __builtin_shufflevector(c0, c1, 0,1,2,3,4,5,6,7,8,9,10,11,12,13,14,15);
}

// B-matrix fragment, 32x16 bf16 (K x N). B = W^T, W row-major [G][D], so each
// lane reads 16 K-elements of one W row (two b128 loads).
// ISA layout: lanes 0-15 -> N=lane, K=kb..kb+7, kb+16..kb+23; lanes 16-31 -> K +8 / +24.
__device__ __forceinline__ v16bf load_b_frag(const unsigned short* w,
                                             int jrow, int kbase, int lane) {
  int col = lane & 15;
  int ko  = (lane >> 4) * 8;
  const unsigned short* p = w + (size_t)(jrow + col) * DD + kbase + ko;
  v8bf c0 = *(const v8bf*)(p);
  v8bf c1 = *(const v8bf*)(p + 16);
  return __builtin_shufflevector(c0, c1, 0,1,2,3,4,5,6,7,8,9,10,11,12,13,14,15);
}

__device__ __forceinline__ v8f wmma_bf16(v16bf a, v16bf b, v8f c) {
  return __builtin_amdgcn_wmma_f32_16x16x32_bf16(false, a, false, b,
                                                 (short)0, c, false, false);
}

// --------------------------- prep kernels ----------------------------------

__global__ void k_convert_w(const float* wi, const float* wh,
                            unsigned short* wib, unsigned short* whb) {
  int i = blockIdx.x * blockDim.x + threadIdx.x;
  int stride = gridDim.x * blockDim.x;
  for (; i < GG * DD; i += stride) {
    wib[i] = f2bf(wi[i]);
    whb[i] = f2bf(wh[i]);
  }
}

__global__ void k_embed(const int* __restrict__ ids,
                        const float* __restrict__ tab, unsigned short* X) {
  int row = blockIdx.x;                       // 0..511 = node*8 + child
  long long id = ids[row];
  const float* src = tab + id * (long long)DD;
  unsigned short* dst = X + row * DD;
  for (int c = threadIdx.x; c < DD; c += blockDim.x) dst[c] = f2bf(src[c]);
}

__global__ void k_init(float* Hf, float* Hsum, unsigned short* Hbf) {
  int i = blockIdx.x * blockDim.x + threadIdx.x;   // 64*512 elements
  if (i < 64 * DD) { Hf[i] = 0.f; Hsum[i] = 0.f; Hbf[i] = 0; }
}

// ------------------------ gi = X * Wih^T + b_ih ----------------------------
// grid (Mtiles, 12), 256 thr = 8 waves; wave -> one 16x16 tile, full K=512.
__global__ void k_gi(const unsigned short* __restrict__ X,
                     const unsigned short* __restrict__ Wb,
                     const float* __restrict__ bih, float* __restrict__ GIp) {
  int lane  = threadIdx.x & 31;
  int wave  = threadIdx.x >> 5;
  int mbase = blockIdx.x * 16;
  int jbase = (blockIdx.y * 8 + wave) * 16;   // 0..1520
  v8f acc = {};
  for (int kk = 0; kk < DD; kk += 32) {
    v16bf a = load_a_frag(X, mbase, kk, lane);
    v16bf b = load_b_frag(Wb, jbase, kk, lane);
    acc = wmma_bf16(a, b, acc);
  }
  int col  = jbase + (lane & 15);
  int half = lane >> 4;
  float bias = bih[col];
  for (int r = 0; r < 8; ++r) {
    int m = mbase + r + 8 * half;
    GIp[(size_t)m * GG + col] = acc[r] + bias;
  }
}

// --------------------------- one GRU time step -----------------------------
// grid (Mtiles, 4), 8 waves; wave owns cols [jbase, jbase+16) of all 3 gates
// for one 16-row tile: 3 WMMA chains sharing the A fragment, then fp32 gate
// math fully in registers. Hbf is double-buffered (src read / dst write).
__global__ void k_step(const unsigned short* __restrict__ Hbsrc,
                       unsigned short* __restrict__ Hbdst,
                       float* __restrict__ Hf, float* __restrict__ Hsum,
                       const float* __restrict__ GIp,
                       const unsigned short* __restrict__ Whb,
                       const float* __restrict__ bhh, int t) {
  int lane  = threadIdx.x & 31;
  int wave  = threadIdx.x >> 5;
  int mbase = blockIdx.x * 16;
  int jbase = (blockIdx.y * 8 + wave) * 16;   // 0..496 (within D)
  v8f cr = {}, cz = {}, cn = {};
  for (int kk = 0; kk < DD; kk += 32) {
    v16bf a  = load_a_frag(Hbsrc, mbase, kk, lane);
    v16bf br = load_b_frag(Whb,          jbase, kk, lane);
    v16bf bz = load_b_frag(Whb,   DD   + jbase, kk, lane);
    v16bf bn = load_b_frag(Whb, 2*DD   + jbase, kk, lane);
    cr = wmma_bf16(a, br, cr);
    cz = wmma_bf16(a, bz, cz);
    cn = wmma_bf16(a, bn, cn);
  }
  int col  = jbase + (lane & 15);
  int half = lane >> 4;
  float bhr = bhh[col], bhz = bhh[DD + col], bhn = bhh[2 * DD + col];
  for (int r = 0; r < 8; ++r) {
    int m = mbase + r + 8 * half;                 // node row
    const float* gi = GIp + (size_t)(m * 8 + t) * GG;
    float xr = gi[col]          + cr[r] + bhr;
    float xz = gi[DD + col]     + cz[r] + bhz;
    float hn =                    cn[r] + bhn;
    float rg = 1.f / (1.f + __expf(-xr));
    float zg = 1.f / (1.f + __expf(-xz));
    float ng = tanhf(gi[2 * DD + col] + rg * hn);
    float hp = Hf[m * DD + col];
    float hv = (1.f - zg) * ng + zg * hp;
    Hf[m * DD + col]    = hv;                     // lane-owned, in-place safe
    Hsum[m * DD + col] += hv;
    Hbdst[m * DD + col] = f2bf(hv);
  }
}

// -------------------- level transition (children -> parents) ---------------
// X_next = mean_t(h) in bf16 (zero-padded to xrows); h0_next = child-mean of
// final h into the *other* fp32 H buffer (rows >= nnext zeroed); Hsum reset.
__global__ void k_trans(const float* __restrict__ HfSrc, float* __restrict__ HfDst,
                        unsigned short* __restrict__ Hbf0, float* __restrict__ Hsum,
                        unsigned short* __restrict__ X,
                        int nprev, int nnext, int xrows) {
  int row = blockIdx.x;                          // 0..63
  for (int col = threadIdx.x; col < DD; col += blockDim.x) {
    float xs = Hsum[row * DD + col] * 0.125f;
    if (row < xrows)
      X[row * DD + col] = (row < nprev) ? f2bf(xs) : (unsigned short)0;
    Hsum[row * DD + col] = 0.f;
    if (row < 16) {
      float h0 = 0.f;
      if (row < nnext) {
        for (int c = 0; c < 8; ++c) h0 += HfSrc[(row * 8 + c) * DD + col];
        h0 *= 0.125f;
      }
      HfDst[row * DD + col] = h0;
      Hbf0[row * DD + col]  = f2bf(h0);
    }
  }
}

__global__ void k_final(const float* __restrict__ Hsum, float* __restrict__ out) {
  int i = blockIdx.x * blockDim.x + threadIdx.x;
  if (i < DD) out[i] = Hsum[i] * 0.125f;         // root node's out.mean
}

// ---------------------------------------------------------------------------

extern "C" void kernel_launch(void* const* d_in, const int* in_sizes, int n_in,
                              void* d_out, int out_size, void* d_ws, size_t ws_size,
                              hipStream_t stream) {
  const int*   leaf_ids = (const int*)d_in[0];
  const float* embed    = (const float*)d_in[2];
  const float* Wih      = (const float*)d_in[3];
  const float* Whh      = (const float*)d_in[4];
  const float* bih      = (const float*)d_in[5];
  const float* bhh      = (const float*)d_in[6];

  char* ws = (char*)d_ws;
  size_t o = 0;
  unsigned short* WihB = (unsigned short*)(ws + o); o += (size_t)GG * DD * 2;
  unsigned short* WhhB = (unsigned short*)(ws + o); o += (size_t)GG * DD * 2;
  unsigned short* X    = (unsigned short*)(ws + o); o += (size_t)512 * DD * 2;
  float*          GIp  = (float*)(ws + o);          o += (size_t)512 * GG * 4;
  float*          Hf0  = (float*)(ws + o);          o += (size_t)64 * DD * 4;
  float*          Hf1  = (float*)(ws + o);          o += (size_t)64 * DD * 4;
  unsigned short* Hb0  = (unsigned short*)(ws + o); o += (size_t)64 * DD * 2;
  unsigned short* Hb1  = (unsigned short*)(ws + o); o += (size_t)64 * DD * 2;
  float*          Hsum = (float*)(ws + o);          o += (size_t)64 * DD * 4;
  (void)ws_size; (void)in_sizes; (void)n_in; (void)out_size;

  k_convert_w<<<768, 256, 0, stream>>>(Wih, Whh, WihB, WhhB);
  k_embed<<<512, 256, 0, stream>>>(leaf_ids, embed, X);
  k_init<<<128, 256, 0, stream>>>(Hf0, Hsum, Hb0);

  // ---- level 3: 64 nodes, gi rows = 512 ----
  k_gi<<<dim3(32, 12), 256, 0, stream>>>(X, WihB, bih, GIp);
  for (int t = 0; t < 8; ++t) {
    const unsigned short* s = (t & 1) ? Hb1 : Hb0;
    unsigned short*       d = (t & 1) ? Hb0 : Hb1;
    k_step<<<dim3(4, 4), 256, 0, stream>>>(s, d, Hf0, Hsum, GIp, WhhB, bhh, t);
  }
  k_trans<<<64, 256, 0, stream>>>(Hf0, Hf1, Hb0, Hsum, X, 64, 8, 64);

  // ---- level 2: 8 nodes, gi rows = 64 ----
  k_gi<<<dim3(4, 12), 256, 0, stream>>>(X, WihB, bih, GIp);
  for (int t = 0; t < 8; ++t) {
    const unsigned short* s = (t & 1) ? Hb1 : Hb0;
    unsigned short*       d = (t & 1) ? Hb0 : Hb1;
    k_step<<<dim3(1, 4), 256, 0, stream>>>(s, d, Hf1, Hsum, GIp, WhhB, bhh, t);
  }
  k_trans<<<64, 256, 0, stream>>>(Hf1, Hf0, Hb0, Hsum, X, 8, 1, 16);

  // ---- level 1: root, gi rows = 16 (padded) ----
  k_gi<<<dim3(1, 12), 256, 0, stream>>>(X, WihB, bih, GIp);
  for (int t = 0; t < 8; ++t) {
    const unsigned short* s = (t & 1) ? Hb1 : Hb0;
    unsigned short*       d = (t & 1) ? Hb0 : Hb1;
    k_step<<<dim3(1, 4), 256, 0, stream>>>(s, d, Hf0, Hsum, GIp, WhhB, bhh, t);
  }
  k_final<<<2, 256, 0, stream>>>(Hsum, (float*)d_out);
}
